// DecoderWithAttention_82506321756306
// MI455X (gfx1250) — compile-verified, matrix operand order checked by reading, not run on hardware
//
#include <hip/hip_runtime.h>

// ---------------------------------------------------------------------------
// Types for WMMA bf16 (gfx1250): 16x16x32, A/B = 16 bf16 per lane, C = 8 f32.
// ---------------------------------------------------------------------------
typedef __attribute__((ext_vector_type(16))) __bf16 v16bf;
typedef __attribute__((ext_vector_type(8)))  __bf16 bf16x8;
typedef __attribute__((ext_vector_type(8)))  float  v8f;

// Problem dimensions (from reference)
static constexpr int Bn   = 100;
static constexpr int Rr   = 15;
static constexpr int Ff   = 4096;
static constexpr int Vv   = 10000;
static constexpr int Ee   = 1024;
static constexpr int Dd   = 1024;
static constexpr int Aa   = 512;
static constexpr int Tt   = 500;
static constexpr int LINn = 1024;
static constexpr int NTn  = 5;
static constexpr int Ll   = 52;
// Padded dims
static constexpr int BP   = 112;    // B padded to 16
static constexpr int BRP  = 1504;   // B*R padded to 16
static constexpr int STP  = 512;    // B*NT padded
static constexpr int TPAD = 512;    // topic K padded (500 -> 512)
// M-tiles per wave: batch GEMMs have exactly 7 M-tiles (BP/16) -> one B-fragment
// load feeds 7 WMMAs; weights stream exactly once per GEMM.
static constexpr int MT   = 7;

__device__ __forceinline__ float sigf(float x) { return 1.0f / (1.0f + __expf(-x)); }

// ---------------------------------------------------------------------------
// WMMA GEMM: C(M,N) += A(M,K) * W(N,K)^T ; A,W bf16 (padded, lda/ldw mult of 8,
// K mult of 32); C fp32 with row guard (r < Mvalid). 8 waves/block, each wave
// owns one 16-wide N tile and MT=7 stacked 16-tall M tiles (56 acc VGPRs).
// ---------------------------------------------------------------------------
__global__ __launch_bounds__(256) void k_gemm(
    float* __restrict__ C, int ldc,
    const __bf16* __restrict__ A, int lda,
    const __bf16* __restrict__ W, int ldw,
    int K, int Mvalid, int N, int MtilesTot)
{
  const int lane = threadIdx.x & 31;
  const int wave = threadIdx.x >> 5;
  const int tn   = blockIdx.x * 8 + wave;      // 16-wide N tile index
  if (tn * 16 >= N) return;                    // whole-wave early out
  const int tm0  = blockIdx.y * MT;
  const int half = lane >> 4;                  // 0: K 0-7/16-23, 1: K 8-15/24-31
  const int l15  = lane & 15;
  const int col  = tn * 16 + l15;

  const __bf16* wRow = W + (size_t)(tn * 16 + l15) * ldw + half * 8;

  const __bf16* aRow[MT];
  v8f acc[MT];
#pragma unroll
  for (int i = 0; i < MT; ++i) {
    const int tmg = tm0 + i;
    const int tmc = (tmg < MtilesTot) ? tmg : (MtilesTot - 1);  // clamp (compute-only)
    aRow[i] = A + (size_t)(tmc * 16 + l15) * lda + half * 8;
#pragma unroll
    for (int v = 0; v < 8; ++v) {
      int r = tmg * 16 + half * 8 + v;
      acc[i][v] = (tmg < MtilesTot && r < Mvalid) ? C[(size_t)r * ldc + col] : 0.0f;
    }
  }

  for (int k = 0; k < K; k += 32) {
    union { v16bf v; bf16x8 h[2]; } b;
    b.h[0] = *(const bf16x8*)(wRow + k);
    b.h[1] = *(const bf16x8*)(wRow + k + 16);
    __builtin_prefetch(wRow + k + 1024, 0, 1);   // stream weights 2KB ahead
#pragma unroll
    for (int i = 0; i < MT; ++i) {
      union { v16bf v; bf16x8 h[2]; } a;
      a.h[0] = *(const bf16x8*)(aRow[i] + k);
      a.h[1] = *(const bf16x8*)(aRow[i] + k + 16);
      // (neg_a, A, neg_b, B, c_mod, C, reuse_a, reuse_b)
      acc[i] = __builtin_amdgcn_wmma_f32_16x16x32_bf16(false, a.v, false, b.v,
                                                       (short)0, acc[i], false, false);
    }
  }

#pragma unroll
  for (int i = 0; i < MT; ++i) {
    const int tmg = tm0 + i;
#pragma unroll
    for (int v = 0; v < 8; ++v) {
      int r = tmg * 16 + half * 8 + v;
      if (tmg < MtilesTot && r < Mvalid) C[(size_t)r * ldc + col] = acc[i][v];
    }
  }
}

// fp32 -> bf16 convert with zero padding: dst is (drows x dld), src is (rows x cols)
__global__ void k_cvt(__bf16* __restrict__ dst, const float* __restrict__ src,
                      int rows, int cols, int dld, int drows)
{
  size_t idx = (size_t)blockIdx.x * blockDim.x + threadIdx.x;
  size_t tot = (size_t)drows * dld;
  if (idx >= tot) return;
  int r = (int)(idx / dld), c = (int)(idx % dld);
  float v = (r < rows && c < cols) ? src[(size_t)r * cols + c] : 0.0f;
  dst[idx] = (__bf16)v;
}

// C[m*ldc+n] = b1[n] (+ b2[n])
__global__ void k_bias2(float* __restrict__ C, int ldc,
                        const float* __restrict__ b1, const float* __restrict__ b2,
                        int M, int N)
{
  size_t idx = (size_t)blockIdx.x * blockDim.x + threadIdx.x;
  if (idx >= (size_t)M * N) return;
  int m = (int)(idx / N), n = (int)(idx % N);
  float v = b1[n];
  if (b2) v += b2[n];
  C[(size_t)m * ldc + n] = v;
}

// img_mean (bf16): mean over R regions
__global__ void k_mean(__bf16* __restrict__ dst, const float* __restrict__ img)
{
  int idx = blockIdx.x * blockDim.x + threadIdx.x;
  if (idx >= Bn * Ff) return;
  int s = idx / Ff, f = idx % Ff;
  float acc = 0.0f;
  for (int r = 0; r < Rr; ++r) acc += img[((size_t)(s * Rr + r)) * Ff + f];
  dst[(size_t)s * Ff + f] = (__bf16)(acc * (1.0f / Rr));
}

// conv2d(1->6, k=(15,26), stride 2) of lt_out; emit topics[:NT] as padded bf16
__global__ void k_conv_topics(__bf16* __restrict__ topb, const float* __restrict__ lt,
                              const float* __restrict__ cw, const float* __restrict__ cb)
{
  int idx = blockIdx.x * blockDim.x + threadIdx.x;
  if (idx >= Bn * NTn * Tt) return;
  int w = idx % Tt, st = idx / Tt;
  int t = st % NTn, s = st / NTn;
  float acc = cb[t];
  for (int r = 0; r < Rr; ++r) {
    const float* row = lt + (size_t)(s * Rr + r) * LINn + 2 * w;
    const float* kw  = cw + (size_t)(t * Rr + r) * 26;
    for (int k = 0; k < 26; ++k) acc += row[k] * kw[k];
  }
  topb[(size_t)st * TPAD + w] = (__bf16)acc;
}

// gather last-word embeddings to bf16
__global__ void k_gather_emb(__bf16* __restrict__ dst, const float* __restrict__ emb,
                             const int* __restrict__ caps, const int* __restrict__ lens)
{
  size_t idx = (size_t)blockIdx.x * blockDim.x + threadIdx.x;
  if (idx >= (size_t)Bn * NTn * Ee) return;
  int e = (int)(idx % Ee); int st = (int)(idx / Ee);
  int len = lens[st];
  int w = caps[(size_t)st * Ll + (len - 1)];
  dst[(size_t)st * Ee + e] = (__bf16)emb[(size_t)w * Ee + e];
}

// LSTM pointwise: gates (BP x 4D), c in/out fp32, h out bf16
__global__ void k_lstm_pw(__bf16* __restrict__ hb, float* __restrict__ c,
                          const float* __restrict__ g)
{
  int idx = blockIdx.x * blockDim.x + threadIdx.x;
  if (idx >= BP * Dd) return;
  int m = idx / Dd, d = idx % Dd;
  const float* gm = g + (size_t)m * 4 * Dd;
  float gi = sigf(gm[d]);
  float gf = sigf(gm[Dd + d]);
  float gg = tanhf(gm[2 * Dd + d]);
  float go = sigf(gm[3 * Dd + d]);
  float cn = gf * c[idx] + gi * gg;
  c[idx] = cn;
  hb[idx] = (__bf16)(go * tanhf(cn));
}

// scores[s,r] = b_full + sum_a relu(att1 + h1proj + topicproj) * W_full[a]
__global__ void k_scores(float* __restrict__ sc, const float* __restrict__ att1,
                         const float* __restrict__ hp, const float* __restrict__ tp,
                         const float* __restrict__ wf, const float* __restrict__ bf,
                         int t)
{
  int idx = blockIdx.x * blockDim.x + threadIdx.x;
  if (idx >= Bn * Rr) return;
  int s = idx / Rr, r = idx % Rr;
  const float* a1 = att1 + (size_t)(s * Rr + r) * Aa;
  const float* h1 = hp + (size_t)s * Aa;
  const float* tpp = tp + (size_t)(s * NTn + t) * TPAD;
  float acc = bf[0];
  for (int a = 0; a < Aa; ++a)
    acc += fmaxf(a1[a] + h1[a] + tpp[a], 0.0f) * wf[a];
  sc[s * 16 + r] = acc;
}

// softmax over R, awe = alpha @ img (fp32 precision), write bf16
__global__ __launch_bounds__(256) void k_softmax_awe(
    __bf16* __restrict__ aweb, const float* __restrict__ sc,
    const float* __restrict__ img)
{
  int s = blockIdx.x;
  __shared__ float alpha[Rr];
  if (threadIdx.x == 0) {
    float mx = -1e30f;
    for (int r = 0; r < Rr; ++r) mx = fmaxf(mx, sc[s * 16 + r]);
    float sum = 0.0f;
    for (int r = 0; r < Rr; ++r) { float e = __expf(sc[s * 16 + r] - mx); alpha[r] = e; sum += e; }
    float inv = 1.0f / sum;
    for (int r = 0; r < Rr; ++r) alpha[r] *= inv;
  }
  __syncthreads();
  for (int f = threadIdx.x; f < Ff; f += blockDim.x) {
    float acc = 0.0f;
    for (int r = 0; r < Rr; ++r) acc += alpha[r] * img[(size_t)(s * Rr + r) * Ff + f];
    aweb[(size_t)s * Ff + f] = (__bf16)acc;
  }
}

// integer outputs 3 & 4 (written as float values)
__global__ void k_emit_ints(float* __restrict__ oc, float* __restrict__ ol,
                            const int* __restrict__ caps, const int* __restrict__ lens)
{
  int idx = blockIdx.x * blockDim.x + threadIdx.x;
  if (idx < Bn * NTn * Ll) oc[idx] = (float)caps[idx];
  if (idx < Bn * NTn)      ol[idx] = (float)(lens[idx] - 1);
}

// ---------------------------------------------------------------------------
extern "C" void kernel_launch(void* const* d_in, const int* in_sizes, int n_in,
                              void* d_out, int out_size, void* d_ws, size_t ws_size,
                              hipStream_t stream)
{
  (void)in_sizes; (void)n_in; (void)out_size; (void)ws_size;
  // inputs (setup_inputs dict order)
  const float* img    = (const float*)d_in[0];
  const int*   caps   = (const int*)d_in[1];
  const int*   lens   = (const int*)d_in[2];
  const float* W_lt   = (const float*)d_in[3];
  const float* b_lt   = (const float*)d_in[4];
  const float* conv_w = (const float*)d_in[5];
  const float* conv_b = (const float*)d_in[6];
  const float* W_feat = (const float*)d_in[7];
  const float* b_feat = (const float*)d_in[8];
  const float* W_dec  = (const float*)d_in[9];
  const float* b_dec  = (const float*)d_in[10];
  const float* W_top  = (const float*)d_in[11];
  const float* b_top  = (const float*)d_in[12];
  const float* W_full = (const float*)d_in[13];
  const float* b_full = (const float*)d_in[14];
  const float* emb    = (const float*)d_in[15];
  const float* W_ih1  = (const float*)d_in[16];
  const float* W_hh1  = (const float*)d_in[17];
  const float* b_ih1  = (const float*)d_in[18];
  const float* b_hh1  = (const float*)d_in[19];
  const float* W_ih2  = (const float*)d_in[20];
  const float* W_hh2  = (const float*)d_in[21];
  const float* b_ih2  = (const float*)d_in[22];
  const float* b_hh2  = (const float*)d_in[23];
  const float* W_fc1  = (const float*)d_in[24];
  const float* b_fc1  = (const float*)d_in[25];
  const float* W_fc   = (const float*)d_in[26];
  const float* b_fc   = (const float*)d_in[27];

  float* out_preds  = (float*)d_out;                         // (B,NT,V)
  float* out_preds1 = out_preds  + (size_t)Bn * NTn * Vv;    // (B,NT,V)
  float* out_caps   = out_preds1 + (size_t)Bn * NTn * Vv;    // (B,NT,L)
  float* out_lens   = out_caps   + (size_t)Bn * NTn * Ll;    // (B,NT)

  // ---- workspace carve-up (256B aligned) ----
  size_t off = 0;
  auto alloc = [&](size_t bytes) {
    char* p = (char*)d_ws + off;
    off += (bytes + 255) & ~(size_t)255;
    return p;
  };
  // bf16 weights (~164MB: fits global L2 after conversion)
  __bf16* Wih1b  = (__bf16*)alloc((size_t)4096 * 6144 * 2);
  __bf16* Whh1b  = (__bf16*)alloc((size_t)4096 * 1024 * 2);
  __bf16* Wih2b  = (__bf16*)alloc((size_t)4096 * 5120 * 2);
  __bf16* Whh2b  = (__bf16*)alloc((size_t)4096 * 1024 * 2);
  __bf16* Wfc1b  = (__bf16*)alloc((size_t)Vv * 1024 * 2);
  __bf16* Wfcb   = (__bf16*)alloc((size_t)Vv * 1024 * 2);
  __bf16* Wltb   = (__bf16*)alloc((size_t)1024 * 4096 * 2);
  __bf16* Wfeatb = (__bf16*)alloc((size_t)512 * 4096 * 2);
  __bf16* Wdecb  = (__bf16*)alloc((size_t)512 * 1024 * 2);
  __bf16* Wtopb  = (__bf16*)alloc((size_t)512 * TPAD * 2);
  size_t act_base = off;
  // activations
  __bf16* imgb   = (__bf16*)alloc((size_t)BRP * Ff * 2);
  float*  ltout  = (float*) alloc((size_t)BRP * LINn * 4);
  float*  att1   = (float*) alloc((size_t)BRP * Aa * 4);
  __bf16* topb   = (__bf16*)alloc((size_t)STP * TPAD * 2);
  float*  tproj  = (float*) alloc((size_t)STP * TPAD * 4);
  __bf16* meanb  = (__bf16*)alloc((size_t)BP * Ff * 2);
  __bf16* embb   = (__bf16*)alloc((size_t)BP * NTn * Ee * 2);
  float*  gates1 = (float*) alloc((size_t)BP * 4 * Dd * 4);
  float*  gates2 = (float*) alloc((size_t)BP * 4 * Dd * 4);
  float*  c1     = (float*) alloc((size_t)BP * Dd * 4);
  float*  c2     = (float*) alloc((size_t)BP * Dd * 4);
  __bf16* h1b    = (__bf16*)alloc((size_t)BP * Dd * 2);
  __bf16* h2b    = (__bf16*)alloc((size_t)BP * Dd * 2);
  float*  h1proj = (float*) alloc((size_t)BP * Aa * 4);
  __bf16* aweb   = (__bf16*)alloc((size_t)BP * Ff * 2);
  float*  sc     = (float*) alloc((size_t)BP * 16 * 4);
  size_t act_bytes = off - act_base;

  // zero activation region each launch (deterministic h0/c0 = 0, padded rows = 0)
  hipMemsetAsync((char*)d_ws + act_base, 0, act_bytes, stream);

  const int TB = 256;
  auto g1 = [&](size_t n) { return dim3((unsigned)((n + TB - 1) / TB)); };
  auto cvt = [&](__bf16* dst, const float* src, int rows, int cols, int dld, int drows) {
    k_cvt<<<g1((size_t)drows * dld), TB, 0, stream>>>(dst, src, rows, cols, dld, drows);
  };
  auto bias = [&](float* C, int ldc, const float* bb1, const float* bb2, int M, int N) {
    k_bias2<<<g1((size_t)M * N), TB, 0, stream>>>(C, ldc, bb1, bb2, M, N);
  };
  auto gemm = [&](float* C, int ldc, const __bf16* Am, int lda,
                  const __bf16* Wm, int ldw, int K, int Mvalid, int Mtiles, int N) {
    dim3 grid((unsigned)((N / 16 + 7) / 8), (unsigned)((Mtiles + MT - 1) / MT));
    k_gemm<<<grid, 256, 0, stream>>>(C, ldc, Am, lda, Wm, ldw, K, Mvalid, N, Mtiles);
  };

  // ---- per-launch weight conversion to bf16 (padded) ----
  cvt(Wih1b,  W_ih1,  4096, 6144, 6144, 4096);
  cvt(Whh1b,  W_hh1,  4096, 1024, 1024, 4096);
  cvt(Wih2b,  W_ih2,  4096, 5120, 5120, 4096);
  cvt(Whh2b,  W_hh2,  4096, 1024, 1024, 4096);
  cvt(Wfc1b,  W_fc1,  Vv,   1024, 1024, Vv);
  cvt(Wfcb,   W_fc,   Vv,   1024, 1024, Vv);
  cvt(Wltb,   W_lt,   1024, 4096, 4096, 1024);
  cvt(Wfeatb, W_feat, 512,  4096, 4096, 512);
  cvt(Wdecb,  W_dec,  512,  1024, 1024, 512);
  cvt(Wtopb,  W_top,  512,  Tt,   TPAD, 512);
  cvt(imgb,   img,    Bn * Rr, Ff, Ff, BRP);

  // ---- one-time per-sample precompute ----
  // lt_out = img @ W_lt^T + b_lt   (1504 x 1024)
  bias(ltout, LINn, b_lt, nullptr, BRP, LINn);
  gemm(ltout, LINn, imgb, Ff, Wltb, Ff, Ff, BRP, BRP / 16, LINn);
  // att1 = img @ W_feat^T + b_feat (1504 x 512)
  bias(att1, Aa, b_feat, nullptr, BRP, Aa);
  gemm(att1, Aa, imgb, Ff, Wfeatb, Ff, Ff, BRP, BRP / 16, Aa);
  // img_mean (bf16)
  k_mean<<<g1((size_t)Bn * Ff), TB, 0, stream>>>(meanb, img);
  // topics (conv) -> bf16 padded (512 x 512)
  k_conv_topics<<<g1((size_t)Bn * NTn * Tt), TB, 0, stream>>>(topb, ltout, conv_w, conv_b);
  // topicproj = topics @ W_top^T + b_top  (512 x 512)
  bias(tproj, TPAD, b_top, nullptr, STP, Aa);
  gemm(tproj, TPAD, topb, TPAD, Wtopb, TPAD, TPAD, STP, STP / 16, Aa);
  // last-word embeddings (bf16)
  k_gather_emb<<<g1((size_t)Bn * NTn * Ee), TB, 0, stream>>>(embb, emb, caps, lens);
  // pass-through integer outputs
  k_emit_ints<<<g1((size_t)Bn * NTn * Ll), TB, 0, stream>>>(out_caps, out_lens, caps, lens);

  // ---- sequential decode: NT steps, batched over B via WMMA GEMMs ----
  for (int t = 0; t < NTn; ++t) {
    // gates1 = [h2, img_mean, emb_t] @ W_ih1^T + h1 @ W_hh1^T + b_ih1 + b_hh1
    bias(gates1, 4 * Dd, b_ih1, b_hh1, BP, 4 * Dd);
    gemm(gates1, 4 * Dd, h2b,             Dd,       Wih1b,        6144, Dd, BP, BP / 16, 4 * Dd);
    gemm(gates1, 4 * Dd, meanb,           Ff,       Wih1b + 1024, 6144, Ff, BP, BP / 16, 4 * Dd);
    gemm(gates1, 4 * Dd, embb + t * Ee,   NTn * Ee, Wih1b + 5120, 6144, Ee, BP, BP / 16, 4 * Dd);
    gemm(gates1, 4 * Dd, h1b,             Dd,       Whh1b,        Dd,   Dd, BP, BP / 16, 4 * Dd);
    k_lstm_pw<<<g1((size_t)BP * Dd), TB, 0, stream>>>(h1b, c1, gates1);

    // attention
    bias(h1proj, Aa, b_dec, nullptr, BP, Aa);
    gemm(h1proj, Aa, h1b, Dd, Wdecb, Dd, Dd, BP, BP / 16, Aa);
    k_scores<<<g1((size_t)Bn * Rr), TB, 0, stream>>>(sc, att1, h1proj, tproj, W_full, b_full, t);
    k_softmax_awe<<<Bn, 256, 0, stream>>>(aweb, sc, img);

    // preds1 = h1 @ W_fc1^T + b_fc1 (direct into d_out, row-guarded)
    bias(out_preds1 + (size_t)t * Vv, NTn * Vv, b_fc1, nullptr, Bn, Vv);
    gemm(out_preds1 + (size_t)t * Vv, NTn * Vv, h1b, Dd, Wfc1b, Dd, Dd, Bn, BP / 16, Vv);

    // gates2 = [awe, h1] @ W_ih2^T + h2 @ W_hh2^T + b_ih2 + b_hh2
    bias(gates2, 4 * Dd, b_ih2, b_hh2, BP, 4 * Dd);
    gemm(gates2, 4 * Dd, aweb, Ff, Wih2b,        5120, Ff, BP, BP / 16, 4 * Dd);
    gemm(gates2, 4 * Dd, h1b,  Dd, Wih2b + 4096, 5120, Dd, BP, BP / 16, 4 * Dd);
    gemm(gates2, 4 * Dd, h2b,  Dd, Whh2b,        Dd,   Dd, BP, BP / 16, 4 * Dd);
    k_lstm_pw<<<g1((size_t)BP * Dd), TB, 0, stream>>>(h2b, c2, gates2);

    // preds = h2 @ W_fc^T + b_fc
    bias(out_preds + (size_t)t * Vv, NTn * Vv, b_fc, nullptr, Bn, Vv);
    gemm(out_preds + (size_t)t * Vv, NTn * Vv, h2b, Dd, Wfcb, Dd, Dd, Bn, BP / 16, Vv);
  }
}
